// CoAttention_72121090834530
// MI455X (gfx1250) — matrix operand order repaired and checked by reading, not compiled, not run on hardware
//
#include <hip/hip_runtime.h>
#include <hip/hip_bf16.h>
#include <math.h>

// CDNA5 / gfx1250: wave32, WMMA 16x16x32 bf16 with f32 accumulate.
typedef __attribute__((ext_vector_type(16))) __bf16 v16bf;
typedef __attribute__((ext_vector_type(8)))  float  v8f;

#define D_FEAT 1024
#define LDS_ROW 1032   // 1024 + 8 pad (ushort units) to spread LDS banks

union FragU { v16bf v; unsigned int u[8]; };

__device__ __forceinline__ unsigned short f2bf(float f) {
  unsigned int u = __float_as_uint(f);
  u += 0x7FFFu + ((u >> 16) & 1u);          // round-to-nearest-even
  return (unsigned short)(u >> 16);
}

__device__ __forceinline__ v8f zero8() {
  v8f z;
#pragma unroll
  for (int i = 0; i < 8; ++i) z[i] = 0.0f;
  return z;
}

// Stage a 16-row x 1024-col fp32 tile into LDS as bf16 (row-major, padded).
__device__ __forceinline__ void stage_tile16(unsigned short* As, const float* src, int ld) {
  for (int idx = threadIdx.x; idx < 16 * 256; idx += 256) {
    int r  = idx >> 8;          // row 0..15
    int c4 = (idx & 255) << 2;  // col (float4 granularity)
    float4 f = *(const float4*)(src + (size_t)r * ld + c4);
    unsigned int a0 = (unsigned int)f2bf(f.x) | ((unsigned int)f2bf(f.y) << 16);
    unsigned int a1 = (unsigned int)f2bf(f.z) | ((unsigned int)f2bf(f.w) << 16);
    unsigned int* d = (unsigned int*)(As + r * LDS_ROW + c4);
    d[0] = a0; d[1] = a1;
  }
}

// A fragment (16x32 bf16): lane = m + 16h; elems 0..15 = K[kb+16h .. +15]
// (contiguous order, matching the B fragment)
__device__ __forceinline__ v16bf load_a_frag(const unsigned short* As, int m, int h, int kb) {
  FragU f;
  const unsigned int* p0 = (const unsigned int*)(As + m * LDS_ROW + kb + 16 * h);
  const unsigned int* p1 = (const unsigned int*)(As + m * LDS_ROW + kb + 16 * h + 8);
#pragma unroll
  for (int j = 0; j < 4; ++j) { f.u[j] = p0[j]; f.u[4 + j] = p1[j]; }
  return f.v;
}

// B fragment (32x16 bf16): lane = t + 16h holds K[kb+16h .. +15] of column t.
// B[k,t] = W[t, kb+k] so this is 16 contiguous bf16 from row t of W (K-major).
__device__ __forceinline__ v16bf load_b_frag(const unsigned short* Wrow, int kb, int h) {
  FragU f;
  const uint4* p = (const uint4*)(Wrow + kb + 16 * h);
  uint4 q0 = p[0], q1 = p[1];
  f.u[0] = q0.x; f.u[1] = q0.y; f.u[2] = q0.z; f.u[3] = q0.w;
  f.u[4] = q1.x; f.u[5] = q1.y; f.u[6] = q1.z; f.u[7] = q1.w;
  return f.v;
}

#define WMMA_BF16(a, b, c) \
  __builtin_amdgcn_wmma_f32_16x16x32_bf16(false, (a), false, (b), (short)0, (c), false, false)

// ---------------------------------------------------------------------------
// fp32 -> bf16 bulk convert (weights); n4 = element_count / 4
__global__ __launch_bounds__(256) void k_cvt_bf16(const float* __restrict__ src,
                                                  unsigned short* __restrict__ dst, int n4) {
  int i = blockIdx.x * 256 + threadIdx.x;
  if (i < n4) {
    float4 f = ((const float4*)src)[i];
    uint2 o;
    o.x = (unsigned int)f2bf(f.x) | ((unsigned int)f2bf(f.y) << 16);
    o.y = (unsigned int)f2bf(f.z) | ((unsigned int)f2bf(f.w) << 16);
    ((uint2*)dst)[i] = o;
  }
}

// ---------------------------------------------------------------------------
// Bias GEMM: Out[r, t] = X[r,:] . W[t,:] + bvec[t]     (M=128, N=K=1024)
__global__ __launch_bounds__(256) void k_bias_gemm(const float* __restrict__ X,
                                                   const unsigned short* __restrict__ Wbf,
                                                   const float* __restrict__ bvec,
                                                   float* __restrict__ Out) {
  __shared__ unsigned short As[16 * LDS_ROW];
  int rowBase = blockIdx.x << 4;
  stage_tile16(As, X + (size_t)rowBase * D_FEAT, D_FEAT);
  __syncthreads();
  int lane = threadIdx.x & 31, wave = threadIdx.x >> 5;
  int m = lane & 15, h = lane >> 4;
  for (int nt = wave; nt < 64; nt += 8) {
    int col = (nt << 4) + m;
    const unsigned short* Wrow = Wbf + (size_t)col * D_FEAT;
    v8f c = zero8();
    for (int kb = 0; kb < D_FEAT; kb += 32)
      c = WMMA_BF16(load_a_frag(As, m, h, kb), load_b_frag(Wrow, kb, h), c);
    float bc = bvec[col];
#pragma unroll
    for (int i = 0; i < 8; ++i)
      Out[(size_t)(rowBase + (h << 3) + i) * D_FEAT + col] = c[i] + bc;
  }
}

// ---------------------------------------------------------------------------
// Fused score GEMM: scores[r] = sum_t v[t] * tanh( X[r,:].W[t,:] + bias[b(r),t] )
// The [rows,1024] projection never hits memory: WMMA tile -> tanh -> dot -> reduce.
__global__ __launch_bounds__(256) void k_score_gemm(const float* __restrict__ X,
                                                    const unsigned short* __restrict__ Wbf,
                                                    const float* __restrict__ bias,
                                                    const float* __restrict__ v,
                                                    float* __restrict__ scores,
                                                    int rowsPerBatch) {
  __shared__ unsigned short As[16 * LDS_ROW];
  __shared__ float sc[16];
  int rowBase = blockIdx.x << 4;
  int batch = rowBase / rowsPerBatch;   // tiles never straddle a batch (64,256 % 16 == 0)
  stage_tile16(As, X + (size_t)rowBase * D_FEAT, D_FEAT);
  if (threadIdx.x < 16) sc[threadIdx.x] = 0.0f;
  __syncthreads();
  int lane = threadIdx.x & 31, wave = threadIdx.x >> 5;
  int m = lane & 15, h = lane >> 4;
  const float* biasRow = bias + (size_t)batch * D_FEAT;
  float acc[8];
#pragma unroll
  for (int i = 0; i < 8; ++i) acc[i] = 0.0f;

  for (int nt = wave; nt < 64; nt += 8) {
    int col = (nt << 4) + m;
    const unsigned short* Wrow = Wbf + (size_t)col * D_FEAT;
    v8f c = zero8();
    for (int kb = 0; kb < D_FEAT; kb += 32) {
      __builtin_prefetch(Wrow + kb + 128, 0, 0);  // speculative global_prefetch
      c = WMMA_BF16(load_a_frag(As, m, h, kb), load_b_frag(Wrow, kb, h), c);
    }
    float bc = biasRow[col], vc = v[col];
#pragma unroll
    for (int i = 0; i < 8; ++i) acc[i] += tanhf(c[i] + bc) * vc;
  }
  // C elem i = (row 8h+i, col lane%16): reduce over the 16 columns of each half-wave
#pragma unroll
  for (int i = 0; i < 8; ++i) {
    float s = acc[i];
    s += __shfl_xor(s, 1); s += __shfl_xor(s, 2);
    s += __shfl_xor(s, 4); s += __shfl_xor(s, 8);
    if (m == 0) atomicAdd(&sc[(h << 3) + i], s);
  }
  __syncthreads();
  if (threadIdx.x < 16) scores[rowBase + threadIdx.x] = sc[threadIdx.x];
}

// ---------------------------------------------------------------------------
// Softmax over N=64 regions (no mask) + weighted sum of feats -> combined[:, colOff..+1024)
__global__ __launch_bounds__(256) void k_ctx_img(const float* __restrict__ scores,
                                                 const float* __restrict__ feats,
                                                 float* __restrict__ combined, int colOff) {
  int b = blockIdx.x, tid = threadIdx.x;
  __shared__ float p[64];
  __shared__ float invZ;
  if (tid < 64) p[tid] = scores[b * 64 + tid];
  __syncthreads();
  if (tid == 0) {
    float mx = p[0];
    for (int n = 1; n < 64; ++n) mx = fmaxf(mx, p[n]);
    float s = 0.0f;
    for (int n = 0; n < 64; ++n) { float e = expf(p[n] - mx); p[n] = e; s += e; }
    invZ = 1.0f / s;
  }
  __syncthreads();
  int d = tid << 2;
  float4 acc = {0.f, 0.f, 0.f, 0.f};
  const float* base = feats + (size_t)b * 64 * D_FEAT + d;
  for (int n = 0; n < 64; ++n) {
    float w = p[n] * invZ;
    float4 f = *(const float4*)(base + (size_t)n * D_FEAT);
    acc.x += w * f.x; acc.y += w * f.y; acc.z += w * f.z; acc.w += w * f.w;
  }
  *(float4*)(combined + (size_t)b * 4096 + colOff + d) = acc;
}

// Masked softmax over L=256 (softmax*mask/renorm == exp*mask/sum(exp*mask)) + weighted sum
__global__ __launch_bounds__(256) void k_ctx_txt(const float* __restrict__ scores,
                                                 const float* __restrict__ feats,
                                                 const float* __restrict__ mask,
                                                 float* __restrict__ combined, int colOff) {
  int b = blockIdx.x, tid = threadIdx.x;
  __shared__ float p[256], red[256];
  float s  = scores[b * 256 + tid];
  float mv = mask[b * 256 + tid];
  red[tid] = s; __syncthreads();
  for (int st = 128; st > 0; st >>= 1) {
    if (tid < st) red[tid] = fmaxf(red[tid], red[tid + st]);
    __syncthreads();
  }
  float mx = red[0]; __syncthreads();
  float e = expf(s - mx) * mv;
  red[tid] = e; __syncthreads();
  for (int st = 128; st > 0; st >>= 1) {
    if (tid < st) red[tid] += red[tid + st];
    __syncthreads();
  }
  float inv = 1.0f / red[0];
  p[tid] = e * inv;
  __syncthreads();
  int d = tid << 2;
  float4 acc = {0.f, 0.f, 0.f, 0.f};
  const float* base = feats + (size_t)b * 256 * D_FEAT + d;
  for (int l = 0; l < 256; ++l) {
    float w = p[l];
    float4 f = *(const float4*)(base + (size_t)l * D_FEAT);
    acc.x += w * f.x; acc.y += w * f.y; acc.z += w * f.z; acc.w += w * f.w;
  }
  *(float4*)(combined + (size_t)b * 4096 + colOff + d) = acc;
}

// ---------------------------------------------------------------------------
// Output GEMM: Out[128,1024] = combined[128,4096] @ W_out^T + b_out, K chunked via LDS
__global__ __launch_bounds__(256) void k_out_gemm(const float* __restrict__ A,
                                                  const unsigned short* __restrict__ Wbf,
                                                  const float* __restrict__ bvec,
                                                  float* __restrict__ Out) {
  __shared__ unsigned short As[16 * LDS_ROW];
  int rowBase = blockIdx.x << 4;
  int lane = threadIdx.x & 31, wave = threadIdx.x >> 5;
  int m = lane & 15, h = lane >> 4;
  v8f c[8];
#pragma unroll
  for (int j = 0; j < 8; ++j) c[j] = zero8();
  for (int kc = 0; kc < 4096; kc += 1024) {
    __syncthreads();
    stage_tile16(As, A + (size_t)rowBase * 4096 + kc, 4096);
    __syncthreads();
#pragma unroll
    for (int j = 0; j < 8; ++j) {
      int col = (((wave << 3) + j) << 4) + m;
      const unsigned short* Wrow = Wbf + (size_t)col * 4096 + kc;
      for (int kb = 0; kb < 1024; kb += 32)
        c[j] = WMMA_BF16(load_a_frag(As, m, h, kb), load_b_frag(Wrow, kb, h), c[j]);
    }
  }
#pragma unroll
  for (int j = 0; j < 8; ++j) {
    int col = (((wave << 3) + j) << 4) + m;
    float bc = bvec[col];
#pragma unroll
    for (int i = 0; i < 8; ++i)
      Out[(size_t)(rowBase + (h << 3) + i) * 1024 + col] = c[j][i] + bc;
  }
}

// ---------------------------------------------------------------------------
extern "C" void kernel_launch(void* const* d_in, const int* in_sizes, int n_in,
                              void* d_out, int out_size, void* d_ws, size_t ws_size,
                              hipStream_t stream) {
  const float* text_feat        = (const float*)d_in[0];
  const float* text_feats       = (const float*)d_in[1];
  const float* img_object_feat  = (const float*)d_in[2];
  const float* img_object_feats = (const float*)d_in[3];
  const float* img_place_feat   = (const float*)d_in[4];
  const float* img_place_feats  = (const float*)d_in[5];
  const float* src_mask         = (const float*)d_in[6];
  const float* v_to  = (const float*)d_in[7];
  const float* v_tp  = (const float*)d_in[8];
  const float* v_io  = (const float*)d_in[9];
  const float* v_ip  = (const float*)d_in[10];
  const float* W_t2o = (const float*)d_in[11];
  const float* W_t2p = (const float*)d_in[12];
  const float* W_o2t = (const float*)d_in[13];
  const float* W_p2t = (const float*)d_in[14];
  const float* W_oo  = (const float*)d_in[15];
  const float* b_oo  = (const float*)d_in[16];
  const float* W_pp  = (const float*)d_in[17];
  const float* b_pp  = (const float*)d_in[18];
  const float* W_tot = (const float*)d_in[19];
  const float* b_tot = (const float*)d_in[20];
  const float* W_tpt = (const float*)d_in[21];
  const float* b_tpt = (const float*)d_in[22];
  const float* W_out = (const float*)d_in[23];
  const float* b_out = (const float*)d_in[24];

  // Workspace layout: 8x (D*D) bf16 + (D*4D) bf16 = 24 MB, then fp32 scratch ~4.5 MB.
  const size_t DD = (size_t)1024 * 1024;
  unsigned short* wbf = (unsigned short*)d_ws;
  unsigned short* Wbf_o2t = wbf + 0 * DD;
  unsigned short* Wbf_p2t = wbf + 1 * DD;
  unsigned short* Wbf_t2o = wbf + 2 * DD;
  unsigned short* Wbf_t2p = wbf + 3 * DD;
  unsigned short* Wbf_tot = wbf + 4 * DD;
  unsigned short* Wbf_tpt = wbf + 5 * DD;
  unsigned short* Wbf_oo  = wbf + 6 * DD;
  unsigned short* Wbf_pp  = wbf + 7 * DD;
  unsigned short* Wbf_out = wbf + 8 * DD;   // 4*DD elems
  float* fws       = (float*)(wbf + 12 * DD);
  float* bias_to   = fws;                    // 128*1024
  float* bias_tp   = bias_to + 131072;
  float* bias_io   = bias_tp + 131072;
  float* bias_ip   = bias_io + 131072;
  float* scores_o  = bias_ip + 131072;       // 128*64
  float* scores_p  = scores_o + 8192;
  float* scores_to = scores_p + 8192;        // 128*256
  float* scores_tp = scores_to + 32768;
  float* combined  = scores_tp + 32768;      // 128*4096

  int n4 = (int)(DD / 4);
  int cvB = n4 / 256;
  k_cvt_bf16<<<cvB, 256, 0, stream>>>(W_o2t, Wbf_o2t, n4);
  k_cvt_bf16<<<cvB, 256, 0, stream>>>(W_p2t, Wbf_p2t, n4);
  k_cvt_bf16<<<cvB, 256, 0, stream>>>(W_t2o, Wbf_t2o, n4);
  k_cvt_bf16<<<cvB, 256, 0, stream>>>(W_t2p, Wbf_t2p, n4);
  k_cvt_bf16<<<cvB, 256, 0, stream>>>(W_tot, Wbf_tot, n4);
  k_cvt_bf16<<<cvB, 256, 0, stream>>>(W_tpt, Wbf_tpt, n4);
  k_cvt_bf16<<<cvB, 256, 0, stream>>>(W_oo,  Wbf_oo,  n4);
  k_cvt_bf16<<<cvB, 256, 0, stream>>>(W_pp,  Wbf_pp,  n4);
  k_cvt_bf16<<<4 * cvB, 256, 0, stream>>>(W_out, Wbf_out, 4 * n4);

  // Per-batch bias vectors (x @ W^T + b), [128,1024] each.
  k_bias_gemm<<<8, 256, 0, stream>>>(text_feat,       Wbf_tot, b_tot, bias_to);
  k_bias_gemm<<<8, 256, 0, stream>>>(text_feat,       Wbf_tpt, b_tpt, bias_tp);
  k_bias_gemm<<<8, 256, 0, stream>>>(img_object_feat, Wbf_oo,  b_oo,  bias_io);
  k_bias_gemm<<<8, 256, 0, stream>>>(img_place_feat,  Wbf_pp,  b_pp,  bias_ip);

  // Fused attention scores (projection never materialized).
  k_score_gemm<<<512,  256, 0, stream>>>(img_object_feats, Wbf_o2t, bias_to, v_to, scores_o,  64);
  k_score_gemm<<<512,  256, 0, stream>>>(img_place_feats,  Wbf_p2t, bias_tp, v_tp, scores_p,  64);
  k_score_gemm<<<2048, 256, 0, stream>>>(text_feats,       Wbf_t2o, bias_io, v_io, scores_to, 256);
  k_score_gemm<<<2048, 256, 0, stream>>>(text_feats,       Wbf_t2p, bias_ip, v_ip, scores_tp, 256);

  // Softmax + weighted sums straight into concatenated layout.
  k_ctx_img<<<128, 256, 0, stream>>>(scores_o,  img_object_feats, combined, 0);
  k_ctx_img<<<128, 256, 0, stream>>>(scores_p,  img_place_feats,  combined, 1024);
  k_ctx_txt<<<128, 256, 0, stream>>>(scores_to, text_feats, src_mask, combined, 2048);
  k_ctx_txt<<<128, 256, 0, stream>>>(scores_tp, text_feats, src_mask, combined, 3072);

  // Final projection.
  k_out_gemm<<<8, 256, 0, stream>>>(combined, Wbf_out, b_out, (float*)d_out);
}
